// ResD1EdgeMLP_6176162972420
// MI455X (gfx1250) — matrix operand order repaired and checked by reading, not compiled, not run on hardware
//
#include <hip/hip_runtime.h>
#include <hip/hip_fp16.h>
#include <math.h>

typedef __attribute__((ext_vector_type(16))) _Float16       v16h;
typedef __attribute__((ext_vector_type(8)))  _Float16       v8h;
typedef __attribute__((ext_vector_type(8)))  float          v8f;
typedef __attribute__((ext_vector_type(16))) unsigned short u16x16;

#define BN 1024
#define DN 64
#define HN 128
#define JSPLIT 16
#define JCHUNK (BN / JSPLIT)   // 64 j's per block

// ---------------------------------------------------------------------------
// Phase 0: zero the relu-sum accumulator S
// ---------------------------------------------------------------------------
__global__ void k_zero(float* __restrict__ p, int n) {
    int i = blockIdx.x * blockDim.x + threadIdx.x;
    if (i < n) p[i] = 0.f;
}

// ---------------------------------------------------------------------------
// Phase 1: tiny precompute.
//   U[i,h]  = (x@Wi)[i,h] + (x@Wd)[i,h]            (i-side affine term)
//   V[j,h]  = (x@Wj)[j,h] - (x@Wd)[j,h] + b1[h]    (j-side affine term)
//   R[i,h]  = (x@Wr)[i,h] + br[h]                  (residual branch)
//   tau[i]  = max(softplus(x@Wt+bt), 0.01) + 1
//   xh      = f16(x)   (feeds WMMA A fragments)
// ---------------------------------------------------------------------------
__global__ void __launch_bounds__(HN)
k_pre(const float* __restrict__ x,  const float* __restrict__ W1,
      const float* __restrict__ b1, const float* __restrict__ Wt,
      const float* __restrict__ bt, const float* __restrict__ Wr,
      const float* __restrict__ br,
      float* __restrict__ U, float* __restrict__ V,
      float* __restrict__ R, float* __restrict__ tau,
      _Float16* __restrict__ xh)
{
    __shared__ float xr[DN];
    const int i = blockIdx.x, t = threadIdx.x;
    if (t < DN) {
        float v = x[i * DN + t];
        xr[t] = v;
        xh[i * DN + t] = (_Float16)v;
    }
    __syncthreads();

    const int h = t;
    const float* Wi  = W1;               // W1 rows   0.. 63
    const float* Wj  = W1 + DN * HN;     // W1 rows  64..127
    const float* Wd  = W1 + 2 * DN * HN; // W1 rows 128..191
    float u = 0.f, v = 0.f, r = 0.f;
    for (int k = 0; k < DN; ++k) {
        float xk = xr[k];
        u += xk * (Wi[k * HN + h] + Wd[k * HN + h]);
        v += xk * (Wj[k * HN + h] - Wd[k * HN + h]);
        r += xk * Wr[k * HN + h];
    }
    U[i * HN + h] = u;
    V[i * HN + h] = v + b1[h];
    R[i * HN + h] = r + br[h];

    if (t == 0) {
        float s = 0.f;
        for (int k = 0; k < DN; ++k) s += xr[k] * Wt[k];
        s += bt[0];
        float sp = (s > 20.f) ? s : log1pf(expf(s));
        tau[i] = fmaxf(sp, 0.01f) + 1.f;
    }
}

// ---------------------------------------------------------------------------
// Phase 2: the 17 GFLOP pairwise kernel (f16 WMMA).
//   S[i,h] += sum_j relu( U[i,h] + V[j,h] + (|x_i - x_j| @ Wab)[h] )
// 2 j's per iteration (4 independent WMMA chains, 8 WMMAs/body) and a
// one-iteration software prefetch of xj/V so the loads never stall the body.
// ---------------------------------------------------------------------------
__device__ __forceinline__ v16h habs16(v16h a) {
    u16x16 u;
    __builtin_memcpy(&u, &a, sizeof(u));
    u &= (unsigned short)0x7FFF;          // clear f16 sign bits
    v16h r;
    __builtin_memcpy(&r, &u, sizeof(r));
    return r;
}

__device__ __forceinline__ v16h cat8(v8h lo, v8h hi) {
    return __builtin_shufflevector(lo, hi, 0,1,2,3,4,5,6,7,8,9,10,11,12,13,14,15);
}

// relu: med3(x, 0, +inf) == clamp to [0, inf)
__device__ __forceinline__ float relu1(float x) {
    return __builtin_amdgcn_fmed3f(x, 0.f, __builtin_inff());
}

__global__ void __launch_bounds__(128)
k_pair(const _Float16* __restrict__ xh, const float* __restrict__ Wab,
       const float* __restrict__ U,     const float* __restrict__ V,
       float* __restrict__ S)
{
    const int lane   = threadIdx.x & 31;
    const int wave   = threadIdx.x >> 5;
    const int n      = lane & 15;
    const int hiHalf = (lane & 16) ? 1 : 0;
    const int i0     = blockIdx.x * 16;
    const int j0     = blockIdx.y * JCHUNK;

    const int h0a = (wave * 2 + 0) * 16;   // each wave owns two 16-wide h tiles
    const int h0b = (wave * 2 + 1) * 16;

    // B fragments of Wab (64x16 tile, split into two K=32 halves).
    // 16-bit B layout: lanes 0-15 hold K={0..7,16..23}, lanes 16-31 +8,
    // element e of the v16h <-> K = kb + e + (e>=8 ? 8 : 0), column N = lane%16.
    const int kb = hiHalf ? 8 : 0;
    v16h bA_lo, bA_hi, bB_lo, bB_hi;
    #pragma unroll
    for (int e = 0; e < 16; ++e) {
        const int ke = kb + e + (e >= 8 ? 8 : 0);
        bA_lo[e] = (_Float16)Wab[(ke     ) * HN + h0a + n];
        bA_hi[e] = (_Float16)Wab[(ke + 32) * HN + h0a + n];
        bB_lo[e] = (_Float16)Wab[(ke     ) * HN + h0b + n];
        bB_hi[e] = (_Float16)Wab[(ke + 32) * HN + h0b + n];
    }

    // U fragments in the f32 C/D layout: VGPR r -> M = r + 8*hiHalf, N = lane%16
    const int Mbase = hiHalf ? 8 : 0;
    v8f uA, uB;
    #pragma unroll
    for (int r = 0; r < 8; ++r) {
        uA[r] = U[(i0 + Mbase + r) * HN + h0a + n];
        uB[r] = U[(i0 + Mbase + r) * HN + h0b + n];
    }

    // A-row fragments for this i-tile (16-bit A layout):
    // lane row M = lane%16; lanes 0-15 hold K = {0..7, 16..23}, lanes 16-31 +8.
    const int ka  = hiHalf ? 8 : 0;
    const int row = i0 + n;
    const v8h* xi_p = reinterpret_cast<const v8h*>(xh + row * DN);
    v16h xiLo = cat8(xi_p[(ka     ) >> 3], xi_p[(ka + 16) >> 3]);  // K  0..31
    v16h xiHi = cat8(xi_p[(ka + 32) >> 3], xi_p[(ka + 48) >> 3]);  // K 32..63

    v8f accA = {}, accB = {};

    // ---- software pipeline: prefetch buffers for one j-pair ahead ----
    v8h p0a, p0b, p0c, p0d, p1a, p1b, p1c, p1d;
    float pvA0, pvB0, pvA1, pvB1;
    {
        const v8h* xj0_p = reinterpret_cast<const v8h*>(xh + (j0    ) * DN);
        const v8h* xj1_p = reinterpret_cast<const v8h*>(xh + (j0 + 1) * DN);
        p0a = xj0_p[(ka     ) >> 3]; p0b = xj0_p[(ka + 16) >> 3];
        p0c = xj0_p[(ka + 32) >> 3]; p0d = xj0_p[(ka + 48) >> 3];
        p1a = xj1_p[(ka     ) >> 3]; p1b = xj1_p[(ka + 16) >> 3];
        p1c = xj1_p[(ka + 32) >> 3]; p1d = xj1_p[(ka + 48) >> 3];
        pvA0 = V[(j0    ) * HN + h0a + n];  pvB0 = V[(j0    ) * HN + h0b + n];
        pvA1 = V[(j0 + 1) * HN + h0a + n];  pvB1 = V[(j0 + 1) * HN + h0b + n];
    }

    for (int jj = 0; jj < JCHUNK; jj += 2) {
        // consume current prefetch
        v16h xj0Lo = cat8(p0a, p0b), xj0Hi = cat8(p0c, p0d);
        v16h xj1Lo = cat8(p1a, p1b), xj1Hi = cat8(p1c, p1d);
        const float vA0 = pvA0, vB0 = pvB0, vA1 = pvA1, vB1 = pvB1;

        // issue next iteration's loads (wraps to j0 on last iter; harmless)
        const int jn = (jj + 2 < JCHUNK) ? (j0 + jj + 2) : j0;
        {
            const v8h* xj0_p = reinterpret_cast<const v8h*>(xh + jn * DN);
            const v8h* xj1_p = reinterpret_cast<const v8h*>(xh + (jn + 1) * DN);
            p0a = xj0_p[(ka     ) >> 3]; p0b = xj0_p[(ka + 16) >> 3];
            p0c = xj0_p[(ka + 32) >> 3]; p0d = xj0_p[(ka + 48) >> 3];
            p1a = xj1_p[(ka     ) >> 3]; p1b = xj1_p[(ka + 16) >> 3];
            p1c = xj1_p[(ka + 32) >> 3]; p1d = xj1_p[(ka + 48) >> 3];
            pvA0 = V[jn * HN + h0a + n];        pvB0 = V[jn * HN + h0b + n];
            pvA1 = V[(jn + 1) * HN + h0a + n];  pvB1 = V[(jn + 1) * HN + h0b + n];
        }

        v16h a0Lo = habs16(xiLo - xj0Lo);
        v16h a0Hi = habs16(xiHi - xj0Hi);
        v16h a1Lo = habs16(xiLo - xj1Lo);
        v16h a1Hi = habs16(xiHi - xj1Hi);

        v8f cA0, cB0, cA1, cB1;
        #pragma unroll
        for (int r = 0; r < 8; ++r) {
            cA0[r] = uA[r] + vA0;  cB0[r] = uB[r] + vB0;
            cA1[r] = uA[r] + vA1;  cB1[r] = uB[r] + vB1;
        }

        // 4 independent chains; all K-lo WMMAs first, then K-hi: each
        // dependent pair has 3 WMMAs + VALU in between -> hazards hidden.
        cA0 = __builtin_amdgcn_wmma_f32_16x16x32_f16(false, a0Lo, false, bA_lo,
                                                     (short)0, cA0, false, false);
        cB0 = __builtin_amdgcn_wmma_f32_16x16x32_f16(false, a0Lo, false, bB_lo,
                                                     (short)0, cB0, false, false);
        cA1 = __builtin_amdgcn_wmma_f32_16x16x32_f16(false, a1Lo, false, bA_lo,
                                                     (short)0, cA1, false, false);
        cB1 = __builtin_amdgcn_wmma_f32_16x16x32_f16(false, a1Lo, false, bB_lo,
                                                     (short)0, cB1, false, false);
        cA0 = __builtin_amdgcn_wmma_f32_16x16x32_f16(false, a0Hi, false, bA_hi,
                                                     (short)0, cA0, false, false);
        cB0 = __builtin_amdgcn_wmma_f32_16x16x32_f16(false, a0Hi, false, bB_hi,
                                                     (short)0, cB0, false, false);
        cA1 = __builtin_amdgcn_wmma_f32_16x16x32_f16(false, a1Hi, false, bA_hi,
                                                     (short)0, cA1, false, false);
        cB1 = __builtin_amdgcn_wmma_f32_16x16x32_f16(false, a1Hi, false, bB_hi,
                                                     (short)0, cB1, false, false);

        #pragma unroll
        for (int r = 0; r < 8; ++r) {
            accA[r] += relu1(cA0[r]) + relu1(cA1[r]);
            accB[r] += relu1(cB0[r]) + relu1(cB1[r]);
        }
    }

    #pragma unroll
    for (int r = 0; r < 8; ++r) {
        unsafeAtomicAdd(&S[(i0 + Mbase + r) * HN + h0a + n], accA[r]);
        unsafeAtomicAdd(&S[(i0 + Mbase + r) * HN + h0b + n], accB[r]);
    }
}

// ---------------------------------------------------------------------------
// Phase 3: finalize (tiny).
//   h_mean = (S/B) @ W2 + b2, /tau ; h = relu(h_mean @ Wa + ba)
//   y = h + R ; out = layernorm(y) * gamma + beta
// ---------------------------------------------------------------------------
__global__ void __launch_bounds__(HN)
k_fin(const float* __restrict__ S,  const float* __restrict__ W2,
      const float* __restrict__ b2, const float* __restrict__ Wa,
      const float* __restrict__ ba, const float* __restrict__ R,
      const float* __restrict__ tau,
      const float* __restrict__ gamma, const float* __restrict__ beta,
      float* __restrict__ out)
{
    __shared__ float m[HN], hb[HN], red[HN];
    const int i = blockIdx.x, h = threadIdx.x;

    m[h] = S[i * HN + h] * (1.f / (float)BN);
    __syncthreads();

    float g = b2[h];
    for (int k = 0; k < HN; ++k) g += m[k] * W2[k * HN + h];
    g /= tau[i];
    hb[h] = g;
    __syncthreads();

    float a = ba[h];
    for (int k = 0; k < HN; ++k) a += hb[k] * Wa[k * HN + h];
    const float y = fmaxf(a, 0.f) + R[i * HN + h];

    red[h] = y;
    __syncthreads();
    for (int s = HN / 2; s > 0; s >>= 1) {
        if (h < s) red[h] += red[h + s];
        __syncthreads();
    }
    const float mu = red[0] * (1.f / HN);
    __syncthreads();

    const float d = y - mu;
    red[h] = d * d;
    __syncthreads();
    for (int s = HN / 2; s > 0; s >>= 1) {
        if (h < s) red[h] += red[h + s];
        __syncthreads();
    }
    const float var = red[0] * (1.f / HN);

    out[i * HN + h] = d * rsqrtf(var + 1e-5f) * gamma[h] + beta[h];
}

// ---------------------------------------------------------------------------
extern "C" void kernel_launch(void* const* d_in, const int* in_sizes, int n_in,
                              void* d_out, int out_size, void* d_ws, size_t ws_size,
                              hipStream_t stream) {
    const float* x     = (const float*)d_in[0];
    const float* W1    = (const float*)d_in[1];
    const float* b1    = (const float*)d_in[2];
    const float* W2    = (const float*)d_in[3];
    const float* b2    = (const float*)d_in[4];
    const float* Wt    = (const float*)d_in[5];
    const float* bt    = (const float*)d_in[6];
    const float* Wa    = (const float*)d_in[7];
    const float* ba    = (const float*)d_in[8];
    const float* Wr    = (const float*)d_in[9];
    const float* br    = (const float*)d_in[10];
    const float* gamma = (const float*)d_in[11];
    const float* beta  = (const float*)d_in[12];

    // workspace layout (floats): U | V | R | S | tau | xh(f16)  ~2.23 MB total
    float* ws  = (float*)d_ws;
    float* U   = ws;
    float* V   = U + BN * HN;
    float* R   = V + BN * HN;
    float* S   = R + BN * HN;
    float* tau = S + BN * HN;
    _Float16* xh = (_Float16*)(tau + BN);

    float* out = (float*)d_out;
    const float* Wab = W1 + 3 * DN * HN;   // W1 rows 192..255

    k_zero<<<(BN * HN + 255) / 256, 256, 0, stream>>>(S, BN * HN);
    k_pre <<<BN, HN, 0, stream>>>(x, W1, b1, Wt, bt, Wr, br, U, V, R, tau, xh);
    k_pair<<<dim3(BN / 16, JSPLIT), 128, 0, stream>>>(xh, Wab, U, V, S);
    k_fin <<<BN, HN, 0, stream>>>(S, W2, b2, Wa, ba, R, tau, gamma, beta, out);
}
